// Attention_45363444580959
// MI455X (gfx1250) — matrix operand order repaired and checked by reading, not compiled, not run on hardware
//
#include <hip/hip_runtime.h>
#include <hip/hip_bf16.h>
#include <stdint.h>

// ---------------- problem constants ----------------
#define B_ 4
#define T_ 2048
#define C_ 1024
#define H_ 16
#define D_ 64
#define M_ (B_ * T_)      // 8192
#define N3_ (3 * C_)      // 3072

typedef __attribute__((ext_vector_type(16))) _Float16 v16h;
typedef __attribute__((ext_vector_type(8)))  float    v8f;

struct U32x8 { uint4 a; uint4 b; };

static __device__ __forceinline__ v16h frag16(const _Float16* p0, const _Float16* p1) {
  U32x8 t;
  t.a = *reinterpret_cast<const uint4*>(p0);
  t.b = *reinterpret_cast<const uint4*>(p1);
  return __builtin_bit_cast(v16h, t);
}

static __device__ __forceinline__ v8f vzero8() {
  v8f z;
#pragma unroll
  for (int i = 0; i < 8; ++i) z[i] = 0.0f;
  return z;
}

static __device__ __forceinline__ v8f wmma16(v16h a, v16h b, v8f c) {
  // D = A(16x32 f16) * B(32x16 f16) + C(16x16 f32)
  return __builtin_amdgcn_wmma_f32_16x16x32_f16(false, a, false, b, (short)0, c,
                                                false, false);
}

// ---------------- kernel 1: fp32 -> f16 convert ----------------
__global__ __launch_bounds__(256) void cvt_f32_f16(const float* __restrict__ in,
                                                   _Float16* __restrict__ out, int n4) {
  int i = blockIdx.x * 256 + threadIdx.x;
  if (i >= n4) return;
  float4 v = reinterpret_cast<const float4*>(in)[i];
  _Float16 h[4] = {(_Float16)v.x, (_Float16)v.y, (_Float16)v.z, (_Float16)v.w};
  reinterpret_cast<uint2*>(out)[i] = *reinterpret_cast<uint2*>(h);
}

// ---------------- kernel 2: transpose weight fp32[K,N] -> f16[N,K] ----------------
__global__ __launch_bounds__(256) void transpose_w(const float* __restrict__ in,
                                                   _Float16* __restrict__ out,
                                                   int K, int N) {
  int idx = blockIdx.x * 256 + threadIdx.x;
  if (idx >= K * N) return;
  int n = idx / K;
  int k = idx - n * K;
  out[(size_t)n * K + k] = (_Float16)in[(size_t)k * N + n];
}

// ---------------- kernel 3: QKV GEMM (M=8192, N=3072, K=1024), double-buffered ----------------
// A: hidden_f16 [M, C] row-major.  Bt: w_qkvT [N, C] row-major (K contiguous).
// Output scattered into Q/K/V as [B, H, T, 64] f16, with fp32 bias.
__global__ __launch_bounds__(256) void qkv_gemm(const _Float16* __restrict__ A,
                                                const _Float16* __restrict__ Bt,
                                                const float* __restrict__ bias,
                                                _Float16* __restrict__ qout,
                                                _Float16* __restrict__ kout,
                                                _Float16* __restrict__ vout) {
  __shared__ __align__(16) _Float16 sA[2][128 * 64];  // [m][k]
  __shared__ __align__(16) _Float16 sB[2][128 * 64];  // [n][k]

  const int tid  = threadIdx.x;
  const int lane = tid & 31, wave = tid >> 5;
  const int r = lane & 15, hlf = lane >> 4;
  const int m0 = blockIdx.y * 128, n0 = blockIdx.x * 128;
  const int m0w = (wave >> 2) * 64;       // 0 or 64 (local)
  const int n0w = (wave & 3) * 32;        // 0..96 (local)

  v8f acc[4][2];
#pragma unroll
  for (int i = 0; i < 4; ++i)
#pragma unroll
    for (int j = 0; j < 2; ++j) acc[i][j] = vzero8();

  const int row  = tid >> 1;
  const int colh = (tid & 1) * 32;
  const _Float16* Arow = A + (size_t)(m0 + row) * C_ + colh;
  const _Float16* Brow = Bt + (size_t)(n0 + row) * C_ + colh;

  // prologue: stage chunk 0
  {
    const uint4* ga = reinterpret_cast<const uint4*>(Arow);
    const uint4* gb = reinterpret_cast<const uint4*>(Brow);
    uint4* la = reinterpret_cast<uint4*>(&sA[0][row * 64 + colh]);
    uint4* lb = reinterpret_cast<uint4*>(&sB[0][row * 64 + colh]);
#pragma unroll
    for (int i = 0; i < 4; ++i) { la[i] = ga[i]; lb[i] = gb[i]; }
  }
  __syncthreads();

  const int NIT = C_ / 64;  // 16
  for (int it = 0; it < NIT; ++it) {
    const int buf = it & 1;
    const bool has_next = (it + 1) < NIT;
    uint4 ra[4], rb[4];
    if (has_next) {
      const uint4* ga = reinterpret_cast<const uint4*>(Arow + (it + 1) * 64);
      const uint4* gb = reinterpret_cast<const uint4*>(Brow + (it + 1) * 64);
#pragma unroll
      for (int i = 0; i < 4; ++i) { ra[i] = ga[i]; rb[i] = gb[i]; }
    }

#pragma unroll
    for (int kk = 0; kk < 64; kk += 32) {
      v16h af[4], bf[2];
#pragma unroll
      for (int mi = 0; mi < 4; ++mi) {
        const _Float16* p = &sA[buf][(m0w + mi * 16 + r) * 64 + kk + 8 * hlf];
        af[mi] = frag16(p, p + 16);
      }
#pragma unroll
      for (int ni = 0; ni < 2; ++ni) {
        const _Float16* p = &sB[buf][(n0w + ni * 16 + r) * 64 + kk + 8 * hlf];
        bf[ni] = frag16(p, p + 16);
      }
#pragma unroll
      for (int mi = 0; mi < 4; ++mi)
#pragma unroll
        for (int ni = 0; ni < 2; ++ni)
          acc[mi][ni] = wmma16(af[mi], bf[ni], acc[mi][ni]);
    }

    if (has_next) {
      uint4* la = reinterpret_cast<uint4*>(&sA[buf ^ 1][row * 64 + colh]);
      uint4* lb = reinterpret_cast<uint4*>(&sB[buf ^ 1][row * 64 + colh]);
#pragma unroll
      for (int i = 0; i < 4; ++i) { la[i] = ra[i]; lb[i] = rb[i]; }
    }
    __syncthreads();
  }

  // epilogue: scatter into Q/K/V [B,H,T,64]
#pragma unroll
  for (int mi = 0; mi < 4; ++mi)
#pragma unroll
    for (int ni = 0; ni < 2; ++ni)
#pragma unroll
      for (int v = 0; v < 8; ++v) {
        int gm = m0 + m0w + mi * 16 + 8 * hlf + v;     // 0..8191
        int gn = n0 + n0w + ni * 16 + r;               // 0..3071
        float val = acc[mi][ni][v] + bias[gn];
        int which = gn >> 10;
        int c = gn & (C_ - 1);
        int h = c >> 6, d = c & 63;
        int b = gm >> 11, t = gm & (T_ - 1);
        size_t dst = (((size_t)b * H_ + h) * T_ + t) * D_ + d;
        _Float16* o = (which == 0) ? qout : (which == 1) ? kout : vout;
        o[dst] = (_Float16)val;
      }
}

// ---------------- kernel 4: causal flash attention per (b,h) ----------------
// Q,K,V: [B,H,T,64] f16.  8 waves/block, 128 Q rows per block, 16 per wave.
// Output: attn [B*T, C] f16 (col = h*64+d).
__global__ __launch_bounds__(256) void flash_attn(const _Float16* __restrict__ Q,
                                                  const _Float16* __restrict__ K,
                                                  const _Float16* __restrict__ V,
                                                  _Float16* __restrict__ out) {
  __shared__ __align__(16) _Float16 sK[32 * 64];   // [key][d]
  __shared__ __align__(16) _Float16 sVt[64 * 32];  // [d][key]
  __shared__ __align__(16) _Float16 sP[8][16 * 32];

  const int tid = threadIdx.x;
  const int lane = tid & 31, wave = tid >> 5;
  const int r = lane & 15, hlf = lane >> 4;

  const int bh = blockIdx.y;
  const int b = bh >> 4, h = bh & 15;
  const size_t base = (size_t)bh * T_ * D_;
  const int q0b = blockIdx.x * 128;
  const int q0w = q0b + wave * 16;

  // preload this wave's Q tile (16x64) as two A fragments (d chunks of 32)
  const _Float16* qp = Q + base + (size_t)q0w * D_;
  v16h aq[2];
#pragma unroll
  for (int kc = 0; kc < 2; ++kc) {
    const _Float16* p = qp + r * 64 + kc * 32 + 8 * hlf;
    aq[kc] = frag16(p, p + 16);
  }

  v8f o[4];
#pragma unroll
  for (int i = 0; i < 4; ++i) o[i] = vzero8();
  float mrow[8], lrow[8];
#pragma unroll
  for (int v = 0; v < 8; ++v) { mrow[v] = -1.0e30f; lrow[v] = 0.0f; }

  const int rowk = tid >> 3;          // 0..31
  const int colh = (tid & 7) * 8;     // 0..56 halfs
  const int nkb = q0b + 128;

  for (int kb = 0; kb < nkb; kb += 32) {
    __syncthreads();  // previous-iteration readers done before restage
    {
      const uint4* gk = reinterpret_cast<const uint4*>(K + base + (size_t)(kb + rowk) * D_ + colh);
      uint4* lk = reinterpret_cast<uint4*>(&sK[rowk * 64 + colh]);
      lk[0] = gk[0];
      const uint4* gv = reinterpret_cast<const uint4*>(V + base + (size_t)(kb + rowk) * D_ + colh);
      uint4 v0 = gv[0];
      _Float16 tmp[8];
      *reinterpret_cast<uint4*>(&tmp[0]) = v0;
#pragma unroll
      for (int i = 0; i < 8; ++i) sVt[(colh + i) * 32 + rowk] = tmp[i];
    }
    // prefetch next key block while this one is consumed (global_prefetch_b8)
    if (kb + 32 < nkb) {
      __builtin_prefetch(K + base + (size_t)(kb + 32 + rowk) * D_ + colh, 0, 3);
      __builtin_prefetch(V + base + (size_t)(kb + 32 + rowk) * D_ + colh, 0, 3);
    }
    __syncthreads();

    if (kb <= q0w + 15) {  // this wave has unmasked keys in this block (wave-uniform)
      // S = Q (16x64) @ K^T (64x32), two 16-col accumulators
      v8f s[2]; s[0] = vzero8(); s[1] = vzero8();
#pragma unroll
      for (int nn = 0; nn < 2; ++nn)
#pragma unroll
        for (int kc = 0; kc < 2; ++kc) {
          const _Float16* p = &sK[(nn * 16 + r) * 64 + kc * 32 + 8 * hlf];
          s[nn] = wmma16(aq[kc], frag16(p, p + 16), s[nn]);
        }

      // scale + causal mask (C/D layout: row = 8*hlf+v, col = lane%16)
#pragma unroll
      for (int nn = 0; nn < 2; ++nn)
#pragma unroll
        for (int v = 0; v < 8; ++v) {
          int row = q0w + 8 * hlf + v;
          int key = kb + nn * 16 + r;
          float x = s[nn][v] * 0.125f;       // 64^-0.5
          s[nn][v] = (key > row) ? -1.0e30f : x;
        }

      // online softmax per row (reduce across 16-lane half with shfl_xor)
#pragma unroll
      for (int v = 0; v < 8; ++v) {
        float mx = fmaxf(s[0][v], s[1][v]);
#pragma unroll
        for (int off = 8; off >= 1; off >>= 1) mx = fmaxf(mx, __shfl_xor(mx, off, 32));
        float newm = fmaxf(mrow[v], mx);
        float alpha = __expf(mrow[v] - newm);
        lrow[v] *= alpha;
#pragma unroll
        for (int ni = 0; ni < 4; ++ni) o[ni][v] *= alpha;
        float p0 = __expf(s[0][v] - newm);
        float p1 = __expf(s[1][v] - newm);
        float sum = p0 + p1;
#pragma unroll
        for (int off = 8; off >= 1; off >>= 1) sum += __shfl_xor(sum, off, 32);
        lrow[v] += sum;
        mrow[v] = newm;
        int rloc = 8 * hlf + v;
        sP[wave][rloc * 32 + r]      = (_Float16)p0;
        sP[wave][rloc * 32 + 16 + r] = (_Float16)p1;
      }

      // LDS stores (C/D layout) -> loads (A layout) within this wave
      asm volatile("s_wait_dscnt 0" ::: "memory");

      const _Float16* pp = &sP[wave][r * 32 + 8 * hlf];
      v16h pf = frag16(pp, pp + 16);
#pragma unroll
      for (int ni = 0; ni < 4; ++ni) {
        const _Float16* vp = &sVt[(ni * 16 + r) * 32 + 8 * hlf];
        o[ni] = wmma16(pf, frag16(vp, vp + 16), o[ni]);
      }
    }
  }

  // epilogue: out[b*T + t, h*64 + d] = o / l
#pragma unroll
  for (int ni = 0; ni < 4; ++ni)
#pragma unroll
    for (int v = 0; v < 8; ++v) {
      int t = q0w + 8 * hlf + v;
      int col = h * 64 + ni * 16 + r;
      float val = o[ni][v] / lrow[v];
      out[((size_t)b * T_ + t) * C_ + col] = (_Float16)val;
    }
}

// ---------------- kernel 5: output projection (M=8192, N=1024, K=1024), double-buffered ----------------
__global__ __launch_bounds__(256) void out_proj(const _Float16* __restrict__ A,
                                                const _Float16* __restrict__ Bt,
                                                const float* __restrict__ bias,
                                                float* __restrict__ out) {
  __shared__ __align__(16) _Float16 sA[2][128 * 64];
  __shared__ __align__(16) _Float16 sB[2][128 * 64];

  const int tid  = threadIdx.x;
  const int lane = tid & 31, wave = tid >> 5;
  const int r = lane & 15, hlf = lane >> 4;
  const int m0 = blockIdx.y * 128, n0 = blockIdx.x * 128;
  const int m0w = (wave >> 2) * 64;
  const int n0w = (wave & 3) * 32;

  v8f acc[4][2];
#pragma unroll
  for (int i = 0; i < 4; ++i)
#pragma unroll
    for (int j = 0; j < 2; ++j) acc[i][j] = vzero8();

  const int row  = tid >> 1;
  const int colh = (tid & 1) * 32;
  const _Float16* Arow = A + (size_t)(m0 + row) * C_ + colh;
  const _Float16* Brow = Bt + (size_t)(n0 + row) * C_ + colh;

  {
    const uint4* ga = reinterpret_cast<const uint4*>(Arow);
    const uint4* gb = reinterpret_cast<const uint4*>(Brow);
    uint4* la = reinterpret_cast<uint4*>(&sA[0][row * 64 + colh]);
    uint4* lb = reinterpret_cast<uint4*>(&sB[0][row * 64 + colh]);
#pragma unroll
    for (int i = 0; i < 4; ++i) { la[i] = ga[i]; lb[i] = gb[i]; }
  }
  __syncthreads();

  const int NIT = C_ / 64;
  for (int it = 0; it < NIT; ++it) {
    const int buf = it & 1;
    const bool has_next = (it + 1) < NIT;
    uint4 ra[4], rb[4];
    if (has_next) {
      const uint4* ga = reinterpret_cast<const uint4*>(Arow + (it + 1) * 64);
      const uint4* gb = reinterpret_cast<const uint4*>(Brow + (it + 1) * 64);
#pragma unroll
      for (int i = 0; i < 4; ++i) { ra[i] = ga[i]; rb[i] = gb[i]; }
    }

#pragma unroll
    for (int kk = 0; kk < 64; kk += 32) {
      v16h af[4], bf[2];
#pragma unroll
      for (int mi = 0; mi < 4; ++mi) {
        const _Float16* p = &sA[buf][(m0w + mi * 16 + r) * 64 + kk + 8 * hlf];
        af[mi] = frag16(p, p + 16);
      }
#pragma unroll
      for (int ni = 0; ni < 2; ++ni) {
        const _Float16* p = &sB[buf][(n0w + ni * 16 + r) * 64 + kk + 8 * hlf];
        bf[ni] = frag16(p, p + 16);
      }
#pragma unroll
      for (int mi = 0; mi < 4; ++mi)
#pragma unroll
        for (int ni = 0; ni < 2; ++ni)
          acc[mi][ni] = wmma16(af[mi], bf[ni], acc[mi][ni]);
    }

    if (has_next) {
      uint4* la = reinterpret_cast<uint4*>(&sA[buf ^ 1][row * 64 + colh]);
      uint4* lb = reinterpret_cast<uint4*>(&sB[buf ^ 1][row * 64 + colh]);
#pragma unroll
      for (int i = 0; i < 4; ++i) { la[i] = ra[i]; lb[i] = rb[i]; }
    }
    __syncthreads();
  }

#pragma unroll
  for (int mi = 0; mi < 4; ++mi)
#pragma unroll
    for (int ni = 0; ni < 2; ++ni) {
      int gn = n0 + n0w + ni * 16 + r;
      float bv = bias[gn];
#pragma unroll
      for (int v = 0; v < 8; ++v) {
        int gm = m0 + m0w + mi * 16 + 8 * hlf + v;
        out[(size_t)gm * C_ + gn] = acc[mi][ni][v] + bv;
      }
    }
}

// ---------------- host launcher ----------------
extern "C" void kernel_launch(void* const* d_in, const int* in_sizes, int n_in,
                              void* d_out, int out_size, void* d_ws, size_t ws_size,
                              hipStream_t stream) {
  (void)in_sizes; (void)n_in; (void)out_size; (void)ws_size;
  const float* hidden = (const float*)d_in[0];
  const float* w_qkv  = (const float*)d_in[1];
  const float* b_qkv  = (const float*)d_in[2];
  const float* w_o    = (const float*)d_in[3];
  const float* b_o    = (const float*)d_in[4];
  float* outp = (float*)d_out;

  char* ws = (char*)d_ws;
  const size_t MB = 1024ull * 1024ull;
  _Float16* hid_h   = (_Float16*)(ws + 0);          // 16 MB  [M, C]
  _Float16* wqkvT_h = (_Float16*)(ws + 16 * MB);    //  6 MB  [3C, C]
  _Float16* woT_h   = (_Float16*)(ws + 22 * MB);    //  2 MB  [C, C]
  _Float16* q_h     = (_Float16*)(ws + 24 * MB);    // 16 MB  [B,H,T,64]
  _Float16* k_h     = (_Float16*)(ws + 40 * MB);    // 16 MB
  _Float16* v_h     = (_Float16*)(ws + 56 * MB);    // 16 MB
  _Float16* attn_h  = (_Float16*)(ws + 72 * MB);    // 16 MB  [M, C]

  // 1) convert activations
  {
    int n4 = (M_ * C_) / 4;
    cvt_f32_f16<<<(n4 + 255) / 256, 256, 0, stream>>>(hidden, hid_h, n4);
  }
  // 2) transpose weights to [N][K] f16
  {
    int tot = N3_ * C_;
    transpose_w<<<(tot + 255) / 256, 256, 0, stream>>>(w_qkv, wqkvT_h, C_, N3_);
    tot = C_ * C_;
    transpose_w<<<(tot + 255) / 256, 256, 0, stream>>>(w_o, woT_h, C_, C_);
  }
  // 3) QKV projection
  qkv_gemm<<<dim3(N3_ / 128, M_ / 128), 256, 0, stream>>>(hid_h, wqkvT_h, b_qkv,
                                                          q_h, k_h, v_h);
  // 4) causal flash attention
  flash_attn<<<dim3(T_ / 128, B_ * H_), 256, 0, stream>>>(q_h, k_h, v_h, attn_h);
  // 5) output projection
  out_proj<<<dim3(C_ / 128, M_ / 128), 256, 0, stream>>>(attn_h, woT_h, b_o, outp);
}
